// LocallyConnected1d_28527172780087
// MI455X (gfx1250) — matrix operand order repaired and checked by reading, not compiled, not run on hardware
//
#include <hip/hip_runtime.h>

// ---------- own ext-vector typedefs (trivially copyable, union-safe) ----------
typedef __attribute__((ext_vector_type(16))) __bf16 v16bf;
typedef __attribute__((ext_vector_type(8)))  __bf16 v8bf;
typedef __attribute__((ext_vector_type(8)))  float  v8f;
typedef __attribute__((ext_vector_type(4)))  unsigned int u32x4;
typedef __attribute__((ext_vector_type(4)))  float  f32x4;

#define BATCH  256
#define C_IN   384
#define C_OUT  384
#define WIDTH  1024
#define KC     32              // K chunk (bf16 16x16x32)
#define NKC    (C_IN / KC)     // 12 chunks
#define NTILE  128             // output-channel tile per workgroup
#define PITCH  48              // bf16/row: 32 data + 16 pad, rows 96B (16B aligned)

union Frag16 { u32x4 q[2]; v16bf v; };

__global__ __launch_bounds__(512)
void lc1d_bf16x3_wmma(const float* __restrict__ x,
                      const float* __restrict__ wts,
                      float* __restrict__ out)
{
    // double-buffered hi/lo bf16 planes: 2*(24+24+12+12) KB = 144 KB
    __shared__ __align__(16) __bf16 XsH[2][BATCH][PITCH];
    __shared__ __align__(16) __bf16 XsL[2][BATCH][PITCH];
    __shared__ __align__(16) __bf16 WsH[2][NTILE][PITCH];
    __shared__ __align__(16) __bf16 WsL[2][NTILE][PITCH];

    const int w   = blockIdx.y;              // one GEMM per width position
    const int n0  = blockIdx.x * NTILE;      // output-channel tile base
    const int tid = threadIdx.x;
    const int lane = tid & 31;
    const int wave = tid >> 5;               // 16 waves
    const int waveM = wave & 3;              // 4 waves x 64 rows (M)
    const int waveN = wave >> 2;             // 4 waves x 32 cols (N)
    const int rowSel = lane & 15;
    const int hsel   = lane >> 4;

    // X staging: thread -> one batch row, 16 consecutive c (immediate-offset loads)
    const int xb  = tid >> 1;                // 0..255
    const int xch = (tid & 1) * 16;          // 0 or 16 within chunk
    // W staging: thread -> one o row, 8 consecutive c
    const int wrow = tid >> 2;               // 0..127
    const int wc8  = (tid & 3) * 8;          // 0,8,16,24 within chunk

    const size_t xBstride = (size_t)C_IN * WIDTH;
    const float* xptr = x + (size_t)xb * xBstride + (size_t)xch * WIDTH + w;
    const float* wptr = wts + (size_t)w * (C_OUT * C_IN)
                            + (size_t)(n0 + wrow) * C_IN + wc8;

    v8f acc[4][2];
#pragma unroll
    for (int mi = 0; mi < 4; ++mi)
#pragma unroll
        for (int ni = 0; ni < 2; ++ni)
            acc[mi][ni] = (v8f)0.0f;

    float xr[16];
    f32x4 wr0, wr1;

    auto loadChunk = [&](int kc) {
#pragma unroll
        for (int i = 0; i < 16; ++i)
            xr[i] = xptr[(size_t)(kc + i) * WIDTH];   // one base, imm offsets (i*4KB)
        wr0 = *(const f32x4*)(wptr + kc);
        wr1 = *(const f32x4*)(wptr + kc + 4);
    };

    auto storeChunk = [&](int buf) {
        v8bf h8, l8;
#pragma unroll
        for (int half = 0; half < 2; ++half) {
#pragma unroll
            for (int j = 0; j < 8; ++j) {
                float  f = xr[half * 8 + j];
                __bf16 h = (__bf16)f;
                h8[j] = h;
                l8[j] = (__bf16)(f - (float)h);
            }
            *(v8bf*)&XsH[buf][xb][xch + half * 8] = h8;   // ds_store_b128
            *(v8bf*)&XsL[buf][xb][xch + half * 8] = l8;
        }
#pragma unroll
        for (int j = 0; j < 8; ++j) {
            float  f = (j < 4) ? wr0[j] : wr1[j - 4];
            __bf16 h = (__bf16)f;
            h8[j] = h;
            l8[j] = (__bf16)(f - (float)h);
        }
        *(v8bf*)&WsH[buf][wrow][wc8] = h8;                // ds_store_b128
        *(v8bf*)&WsL[buf][wrow][wc8] = l8;
    };

    auto computeChunk = [&](int buf) {
        // A 16x32 layout: row = lane&15; K segs {hsel*8..+7} and {16+hsel*8..+7}
        Frag16 aH[4], aL[4];
#pragma unroll
        for (int mi = 0; mi < 4; ++mi) {
            const __bf16* pH = &XsH[buf][waveM * 64 + mi * 16 + rowSel][hsel * 8];
            const __bf16* pL = &XsL[buf][waveM * 64 + mi * 16 + rowSel][hsel * 8];
            aH[mi].q[0] = *(const u32x4*)(pH);
            aH[mi].q[1] = *(const u32x4*)(pH + 16);
            aL[mi].q[0] = *(const u32x4*)(pL);
            aL[mi].q[1] = *(const u32x4*)(pL + 16);
        }
        // B 32x16 layout: col = lane&15; K segs {hsel*8..+7} and {16+hsel*8..+7} (B[k][n] = W[n][k])
        Frag16 bH[2], bL[2];
#pragma unroll
        for (int ni = 0; ni < 2; ++ni) {
            const __bf16* pH = &WsH[buf][waveN * 32 + ni * 16 + rowSel][hsel * 8];
            const __bf16* pL = &WsL[buf][waveN * 32 + ni * 16 + rowSel][hsel * 8];
            bH[ni].q[0] = *(const u32x4*)(pH);
            bH[ni].q[1] = *(const u32x4*)(pH + 16);
            bL[ni].q[0] = *(const u32x4*)(pL);
            bL[ni].q[1] = *(const u32x4*)(pL + 16);
        }
        // bf16x3: aH*bH + aH*bL + aL*bH
#pragma unroll
        for (int mi = 0; mi < 4; ++mi)
#pragma unroll
            for (int ni = 0; ni < 2; ++ni) {
                acc[mi][ni] = __builtin_amdgcn_wmma_f32_16x16x32_bf16(
                    false, aH[mi].v, false, bH[ni].v, (short)0, acc[mi][ni], false, false);
                acc[mi][ni] = __builtin_amdgcn_wmma_f32_16x16x32_bf16(
                    false, aH[mi].v, false, bL[ni].v, (short)0, acc[mi][ni], false, false);
                acc[mi][ni] = __builtin_amdgcn_wmma_f32_16x16x32_bf16(
                    false, aL[mi].v, false, bH[ni].v, (short)0, acc[mi][ni], false, false);
            }
    };

    // ---- software pipeline: 1 barrier/chunk, loads(k+1) overlap WMMAs(k) ----
    loadChunk(0);
    storeChunk(0);
    for (int k = 0; k < NKC; ++k) {
        __syncthreads();                 // buf[k&1] visible; buf[(k+1)&1] readers done
        if (k + 1 < NKC) loadChunk((k + 1) * KC);   // long-latency global loads first
        computeChunk(k & 1);
        if (k + 1 < NKC) storeChunk((k + 1) & 1);   // convert + ds_store after WMMAs
    }

    // ---- store: f32 C/D layout — elem r: M = r + 8*hsel, N = lane&15 ----
    const int oCol = n0 + waveN * 32 + rowSel;
#pragma unroll
    for (int mi = 0; mi < 4; ++mi)
#pragma unroll
        for (int ni = 0; ni < 2; ++ni)
#pragma unroll
            for (int r = 0; r < 8; ++r) {
                int b = waveM * 64 + mi * 16 + r + hsel * 8;
                int o = oCol + ni * 16;
                out[(size_t)b * (C_OUT * WIDTH) + (size_t)o * WIDTH + w] = acc[mi][ni][r];
            }
}

extern "C" void kernel_launch(void* const* d_in, const int* in_sizes, int n_in,
                              void* d_out, int out_size, void* d_ws, size_t ws_size,
                              hipStream_t stream)
{
    (void)in_sizes; (void)n_in; (void)d_ws; (void)ws_size; (void)out_size;
    const float* x   = (const float*)d_in[0];
    const float* wts = (const float*)d_in[1];
    float* out = (float*)d_out;

    dim3 grid(C_OUT / NTILE, WIDTH);   // (3, 1024): n-tile fastest -> same-w blocks adjacent
    dim3 block(512);                   // 16 waves
    lc1d_bf16x3_wmma<<<grid, block, 0, stream>>>(x, wts, out);
}
